// ConvHex_58411555226180
// MI455X (gfx1250) — compile-verified, matrix operand order checked by reading, not run on hardware
//
#include <hip/hip_runtime.h>
#include <hip/hip_bf16.h>
#include <math.h>

typedef __attribute__((ext_vector_type(16))) _Float16 v16h;
typedef __attribute__((ext_vector_type(8)))  _Float16 v8h;
typedef __attribute__((ext_vector_type(4)))  _Float16 v4h;
typedef __attribute__((ext_vector_type(8)))  float    v8f;
typedef __attribute__((ext_vector_type(4)))  float    v4f;

#define N_BATCH 16
#define H_IN    209
#define W_IN    133
#define C_IN    64
#define C_OUT   128
#define H_OUT   201
#define W_OUT   129
#define N_TAPS  19
#define K_TOTAL (N_TAPS * C_IN)     // 1216
#define HTILE   8                   // output rows per workgroup
#define ROWS_IN (HTILE + 8)         // 16 input rows per tile
#define COLS_IN 20                  // 16 output cols + KW-1
#define CISTR   72                  // padded f16 stride per (row,col) -> conflict-free b128

// hex_indices(2) tap offsets (dh, dw); compile-time so the unrolled K loop
// folds every address into base + immediate offset.
constexpr int DH[N_TAPS] = {4,3,5,6,5,3,2,1,2,4,6,7,8,7,6,4,2,1,0};
constexpr int DW[N_TAPS] = {2,3,3,2,1,1,2,3,4,4,4,3,2,1,0,0,0,1,2};

// ---------------------------------------------------------------------------
// Repack sparse weights f32[ci*2432 + co*19 + tap] -> f16 Wpack[kblk][co][khalf]
// k = tap*64 + ci ; kblk = k/32 ; khalf = k%32  (matches WMMA 16-bit B layout)
// ---------------------------------------------------------------------------
__global__ void repack_weights(const float* __restrict__ sw,
                               _Float16* __restrict__ wp) {
    int p = blockIdx.x * blockDim.x + threadIdx.x;
    if (p >= K_TOTAL * C_OUT) return;
    int khalf = p & 31;
    int co    = (p >> 5) & 127;
    int kblk  = p >> 12;
    int tap   = kblk >> 1;
    int ci    = ((kblk & 1) << 5) | khalf;
    float v = sw[ci * (N_TAPS * C_OUT) + co * N_TAPS + tap];
    wp[p] = (_Float16)v;
}

// ---------------------------------------------------------------------------
// Implicit-GEMM hex conv: per WG -> 8 rows x 16 cols x 128 co.
// 8 wave32's; each wave owns a 16-channel N slice and 8 M-tiles (rows).
// Fully unrolled K loop: 19 taps x 2 ci-blocks x 8 rows = 304 WMMAs.
// ---------------------------------------------------------------------------
__global__ __launch_bounds__(256, 2)
void hexconv_wmma(const float* __restrict__ x,
                  const _Float16* __restrict__ wp,
                  const float* __restrict__ offs,
                  float* __restrict__ out) {
    __shared__ __align__(16) _Float16 smem[ROWS_IN * COLS_IN * CISTR]; // 46080 B

    const int w0 = blockIdx.x * 16;
    const int h0 = blockIdx.y * HTILE;
    const int n  = blockIdx.z;
    const int tid = threadIdx.x;

    // ---- stage input patch (f32 -> f16) into LDS ----
    // 16 rows * 20 cols * 16 groups of 4 ci = 5120 jobs, 20 per thread
    for (int j = tid; j < ROWS_IN * COLS_IN * 16; j += 256) {
        int row = j / (COLS_IN * 16);
        int rem = j - row * (COLS_IN * 16);
        int col = rem >> 4;
        int q   = rem & 15;
        int gr = h0 + row; gr = gr > (H_IN - 1) ? (H_IN - 1) : gr;  // clamp (masked later)
        int gc = w0 + col; gc = gc > (W_IN - 1) ? (W_IN - 1) : gc;
        const v4f xv = *(const v4f*)(x + (((size_t)n * H_IN + gr) * W_IN + gc) * C_IN + q * 4);
        v4h hv;
        hv.x = (_Float16)xv.x; hv.y = (_Float16)xv.y;
        hv.z = (_Float16)xv.z; hv.w = (_Float16)xv.w;
        *(v4h*)(smem + (row * COLS_IN + col) * CISTR + q * 4) = hv;
    }
    __syncthreads();

    const int lane  = tid & 31;
    const int wave  = tid >> 5;          // 0..7
    const int co0   = wave * 16;
    const int nlo   = lane & 15;
    const int hiSel = (lane >> 4) * 8;   // 0 for lanes 0-15, 8 for lanes 16-31

    // Per-lane bases; all further addressing folds to immediate offsets.
    const _Float16* abase = smem + nlo * CISTR + hiSel;                 // + (dh*20+dw)*72 + cb*32 (+16)
    const _Float16* bbase = wp + ((size_t)(co0 + nlo)) * 32 + hiSel;    // + kblk*4096 (+16)

    v8f acc[HTILE];
    #pragma unroll
    for (int m = 0; m < HTILE; ++m) acc[m] = (v8f){};

    #pragma unroll
    for (int tap = 0; tap < N_TAPS; ++tap) {
        #pragma unroll
        for (int cb = 0; cb < 2; ++cb) {
            const int kblk = tap * 2 + cb;
            // B fragment: two contiguous b128 loads per lane from packed weights
            union { v16h v; v8h h[2]; } B;
            B.h[0] = *(const v8h*)(bbase + (size_t)kblk * (C_OUT * 32));
            B.h[1] = *(const v8h*)(bbase + (size_t)kblk * (C_OUT * 32) + 16);
            const int aoff = (DH[tap] * COLS_IN + DW[tap]) * CISTR + cb * 32;
            #pragma unroll
            for (int m = 0; m < HTILE; ++m) {
                const _Float16* ap = abase + aoff + m * (COLS_IN * CISTR);
                union { v16h v; v8h h[2]; } A;
                A.h[0] = *(const v8h*)(ap);        // K base + 0..7
                A.h[1] = *(const v8h*)(ap + 16);   // K base + 16..23
                acc[m] = __builtin_amdgcn_wmma_f32_16x16x32_f16(
                    false, A.v, false, B.v, (short)0, acc[m], false, false);
            }
        }
    }

    // ---- epilogue: +offset, ELU, hex mask, store ----
    const float off = offs[co0 + nlo];
    #pragma unroll
    for (int m = 0; m < HTILE; ++m) {
        const int h = h0 + m;
        if (h < H_OUT) {
            #pragma unroll
            for (int r = 0; r < 8; ++r) {
                const int wpix = w0 + r + hiSel;   // D-matrix M = r (+8 for upper lanes)
                if (wpix < W_OUT) {
                    float v = acc[m][r] + off;
                    v = v > 0.f ? v : expm1f(v);
                    // hex mask: doubled-height hex distance, radius 64, center (100,64)
                    const int di = h - 100;
                    const int dj = wpix - 64;
                    const int dx = dj < 0 ? -dj : dj;
                    const int dy = di < 0 ? -di : di;
                    const int dist = dx + (dy > dx ? ((dy - dx) >> 1) : 0);
                    const bool inside = (((di + dj) & 1) == 0) && (dist <= 64);
                    v = inside ? v : 0.f;
                    out[(((size_t)n * H_OUT + h) * W_OUT + wpix) * C_OUT + co0 + nlo] = v;
                }
            }
        }
    }
}

extern "C" void kernel_launch(void* const* d_in, const int* in_sizes, int n_in,
                              void* d_out, int out_size, void* d_ws, size_t ws_size,
                              hipStream_t stream) {
    const float* x    = (const float*)d_in[0];
    const float* sw   = (const float*)d_in[1];
    const float* offs = (const float*)d_in[2];
    float* out = (float*)d_out;
    _Float16* wp = (_Float16*)d_ws;   // 1216*128*2 = 311296 B of scratch

    repack_weights<<<(K_TOTAL * C_OUT + 255) / 256, 256, 0, stream>>>(sw, wp);

    dim3 grid((W_OUT + 15) / 16, (H_OUT + HTILE - 1) / HTILE, N_BATCH);
    hexconv_wmma<<<grid, 256, 0, stream>>>(x, wp, offs, out);
}
